// SSDLayer_7713761264352
// MI455X (gfx1250) — matrix operand order repaired and checked by reading, not compiled, not run on hardware
//
#include <hip/hip_runtime.h>
#include <hip/hip_bf16.h>

// ---------------------------------------------------------------------------
// Mamba-2 SSD layer forward for gfx1250 (MI455X).
// Big GEMMs: bf16 WMMA (v_wmma_f32_16x16x32_bf16, f32 accumulate),
//            weights staged to LDS via the Tensor Data Mover (TDM).
// SSD chunk matmuls: fp32 WMMA (v_wmma_f32_16x16x4_f32) for precision.
// ---------------------------------------------------------------------------

typedef __bf16 bf16_t;
typedef __bf16 v8bf  __attribute__((ext_vector_type(8)));
typedef __bf16 v16bf __attribute__((ext_vector_type(16)));
typedef float  v2f   __attribute__((ext_vector_type(2)));
typedef float  v8f   __attribute__((ext_vector_type(8)));
typedef unsigned int v4u __attribute__((ext_vector_type(4)));
typedef int    v4i   __attribute__((ext_vector_type(4)));
typedef int    v8i   __attribute__((ext_vector_type(8)));

#define MROWS   65536   // B*L
#define LSEQ    2048
#define BATCH   32
#define DMODEL  256
#define DINNER  512
#define DSTATE  64
#define NH      8
#define HDIM    64
#define CHUNK_  64
#define NC_     32
#define CONVD   640
#define NPROJ   1160
#define NPROJP  1168    // padded to multiple of 16
#define DFFN    1024
#define KTILE   256     // K staging tile for TDM -> LDS

// ---------------------------------------------------------------------------
// f32 -> bf16 convert with optional row padding (rows >= N written as zero)
// ---------------------------------------------------------------------------
__global__ void cvt_pad_kernel(const float* __restrict__ src, bf16_t* __restrict__ dst,
                               int N, int K, long long total) {
  long long i = (long long)blockIdx.x * 256 + threadIdx.x;
  if (i >= total) return;
  int k = (int)(i % K);
  long long n = i / K;
  dst[i] = (n < N) ? (bf16_t)src[n * K + k] : (bf16_t)0.0f;
}

// ---------------------------------------------------------------------------
// bf16 WMMA GEMM:  C[M,N] = A[M,K] * W[N,K]^T + bias  (both row-major along K)
// Block: 256 threads = 8 waves; tile 64(m) x 64(n).
// Wave (w): m-tile = (w>>1), n-half = (w&1) covering two 16-wide n tiles.
// Weight tile (64 rows x KTILE) staged global->LDS via TENSOR_LOAD_TO_LDS,
// completion via s_wait_tensorcnt; B fragments then read from LDS.
// TDM zero-fills rows beyond Npad (OOB reads return zero) -> ragged edge safe.
// ACT: 0 = none, 1 = hardswish.  OUTBF: 0 -> f32 out, 1 -> bf16 out.
// ---------------------------------------------------------------------------
template <int ACT, int OUTBF>
__global__ void __launch_bounds__(256) gemm_bf16_kernel(
    const bf16_t* __restrict__ A, const bf16_t* __restrict__ W,
    const float* __restrict__ bias, void* __restrict__ Cout,
    int Mr, int Npad, int N, int K, int ldc) {
  __shared__ bf16_t ldsB[64 * KTILE];   // 32 KB

  int lane = threadIdx.x & 31;
  int wave = threadIdx.x >> 5;
  int m0 = blockIdx.y * 64 + (wave >> 1) * 16;
  int nb = blockIdx.x * 64;                 // block n base
  int n0 = nb + (wave & 1) * 32;            // wave n base (two 16-wide tiles)
  int ml = lane & 15;
  int hi = lane >> 4;

  const bf16_t* arow = A + (size_t)(m0 + ml) * K;

  int rows_valid = Npad - nb;
  if (rows_valid > 64) rows_valid = 64;
  if (rows_valid < 0) rows_valid = 0;
  unsigned int lds_base = (unsigned int)(size_t)&ldsB[0];   // LDS byte offset

  v8f acc0 = (v8f)0.0f;
  v8f acc1 = (v8f)0.0f;

  for (int kt = 0; kt < K; kt += KTILE) {
    __syncthreads();   // previous LDS tile fully consumed
    if (wave == 0) {
      // ---- Tensor DMA descriptor (D#), ISA ch.8 ----
      unsigned long long gaddr =
          (unsigned long long)(size_t)(W + (size_t)nb * K + kt);
      v4u g0;
      g0.x = 1u;                                        // count=1, user mode
      g0.y = lds_base;                                  // lds_addr (bytes)
      g0.z = (unsigned int)(gaddr & 0xFFFFFFFFu);       // global_addr[31:0]
      g0.w = (unsigned int)((gaddr >> 32) & 0x1FFFFFFu) // global_addr[56:32]
             | (2u << 30);                              // type=2 ("image")
      v8i g1;
      g1[0] = (1 << 16);                                // data_size=1 (2B)
      g1[1] = (KTILE & 0xFFFF) << 16;                   // tensor_dim0 lo16
      g1[2] = ((KTILE >> 16) & 0xFFFF)                  // tensor_dim0 hi16
              | ((rows_valid & 0xFFFF) << 16);          // tensor_dim1 lo16
      g1[3] = ((rows_valid >> 16) & 0xFFFF)             // tensor_dim1 hi16
              | (KTILE << 16);                          // tile_dim0
      g1[4] = 64;                                       // tile_dim1=64, tile_dim2=0
      g1[5] = K;                                        // tensor_dim0_stride lo32
      g1[6] = 0;                                        // stride hi / dim1_stride lo
      g1[7] = 0;
      v4i z4 = (v4i)0;
      v8i z8 = (v8i)0;
      __builtin_amdgcn_tensor_load_to_lds(g0, g1, z4, z4, z8, 0);
      __builtin_amdgcn_s_wait_tensorcnt(0);
    }
    __syncthreads();   // LDS tile visible to all waves

    for (int kk = 0; kk < KTILE; kk += 32) {
      int k0 = kt + kk;
      if (k0 + 32 < K) __builtin_prefetch(arow + k0 + 32, 0, 1);  // global_prefetch_b8
      // A fragment (16x32 bf16): lane m = ml, K-halves selected by hi
      v8bf a_lo = *(const v8bf*)(arow + k0 + hi * 8);
      v8bf a_hi = *(const v8bf*)(arow + k0 + 16 + hi * 8);
      v16bf afrag = __builtin_shufflevector(a_lo, a_hi,
          0,1,2,3,4,5,6,7,8,9,10,11,12,13,14,15);
      // B fragments from LDS (row-major [64][KTILE])
      const bf16_t* b0p = &ldsB[((wave & 1) * 32 + ml) * KTILE + kk + hi * 8];
      const bf16_t* b1p = b0p + 16 * KTILE;
      v8bf b0lo = *(const v8bf*)b0p;
      v8bf b0hi = *(const v8bf*)(b0p + 16);
      v8bf b1lo = *(const v8bf*)b1p;
      v8bf b1hi = *(const v8bf*)(b1p + 16);
      v16bf bf0 = __builtin_shufflevector(b0lo, b0hi,
          0,1,2,3,4,5,6,7,8,9,10,11,12,13,14,15);
      v16bf bf1 = __builtin_shufflevector(b1lo, b1hi,
          0,1,2,3,4,5,6,7,8,9,10,11,12,13,14,15);
      acc0 = __builtin_amdgcn_wmma_f32_16x16x32_bf16(
          false, afrag, false, bf0, (short)0, acc0, false, false);
      acc1 = __builtin_amdgcn_wmma_f32_16x16x32_bf16(
          false, afrag, false, bf1, (short)0, acc1, false, false);
    }
  }

  // epilogue: two 16x16 tiles per wave
  #pragma unroll
  for (int j = 0; j < 2; ++j) {
    int n = n0 + j * 16 + ml;
    if (n < N) {
      float bz = bias ? bias[n] : 0.0f;
      v8f acc = j ? acc1 : acc0;
      #pragma unroll
      for (int r = 0; r < 8; ++r) {
        int m = m0 + hi * 8 + r;
        float v = acc[r] + bz;
        if (ACT == 1) {  // hardswish
          float t = v + 3.0f;
          t = fminf(fmaxf(t, 0.0f), 6.0f);
          v = v * t * (1.0f / 6.0f);
        }
        if (OUTBF) ((bf16_t*)Cout)[(size_t)m * ldc + n] = (bf16_t)v;
        else       ((float*)Cout)[(size_t)m * ldc + n] = v;
      }
    }
  }
}

// ---------------------------------------------------------------------------
// Causal depthwise conv (K=4) + SiLU + split into xs / B / C
// ---------------------------------------------------------------------------
__global__ void conv_silu_split_kernel(const float* __restrict__ zx,
                                       const float* __restrict__ conv_w,
                                       const float* __restrict__ conv_b,
                                       float* __restrict__ xs,
                                       float* __restrict__ Bm,
                                       float* __restrict__ Cm) {
  long long idx = (long long)blockIdx.x * 256 + threadIdx.x;
  if (idx >= (long long)MROWS * CONVD) return;
  int c = (int)(idx % CONVD);
  long long row = idx / CONVD;
  int l = (int)(row % LSEQ);
  float acc = conv_b[c];
  #pragma unroll
  for (int k = 0; k < 4; ++k) {
    int ls = l - 3 + k;
    if (ls >= 0)
      acc += zx[(row - (long long)(3 - k)) * NPROJP + DINNER + c] * conv_w[c * 4 + k];
  }
  float s = acc / (1.0f + expf(-acc));  // silu
  if (c < DINNER)               xs[row * DINNER + c] = s;
  else if (c < DINNER + DSTATE) Bm[row * DSTATE + (c - DINNER)] = s;
  else                          Cm[row * DSTATE + (c - DINNER - DSTATE)] = s;
}

// ---------------------------------------------------------------------------
// dt softplus, dA = dt*A, per-chunk inclusive cumsum, chunk decay
// one block per (b,h,chunk), 64 threads
// ---------------------------------------------------------------------------
__global__ void dt_scan_kernel(const float* __restrict__ zx,
                               const float* __restrict__ dt_bias,
                               const float* __restrict__ A_log,
                               float* __restrict__ dtsp,
                               float* __restrict__ Acs,
                               float* __restrict__ cdec) {
  int t = threadIdx.x;                 // 0..63
  int bid = blockIdx.x;
  int c = bid % NC_;
  int h = (bid / NC_) % NH;
  int b = bid / (NC_ * NH);
  int l = c * CHUNK_ + t;
  long long row = (long long)b * LSEQ + l;
  float raw = zx[row * NPROJP + (NPROJ - NH) + h] + dt_bias[h];
  float sp = (raw > 20.0f) ? raw : log1pf(expf(raw));   // softplus
  dtsp[row * NH + h] = sp;
  float dA = sp * (-expf(A_log[h]));
  __shared__ float sh[CHUNK_];
  sh[t] = dA;
  __syncthreads();
  float cum = 0.0f;
  for (int i = 0; i <= t; ++i) cum += sh[i];
  long long aidx = (((long long)b * NH + h) * NC_ + c) * CHUNK_ + t;
  Acs[aidx] = cum;
  if (t == CHUNK_ - 1) cdec[((long long)b * NH + h) * NC_ + c] = expf(cum);
}

// ---------------------------------------------------------------------------
// Per-chunk states: states[p,n] = sum_l Xd[l,p]*exp(Acs_last-Acs[l])*B[l,n]
// f32 WMMA 16x16x4; block = 512 thr (16 waves -> 4x4 tiles), grid (b,c,h)
// ---------------------------------------------------------------------------
__global__ void __launch_bounds__(512) ssd_states_kernel(
    const float* __restrict__ xs, const float* __restrict__ Bm,
    const float* __restrict__ dtsp, const float* __restrict__ Acs,
    float* __restrict__ states) {
  int tid = threadIdx.x, lane = tid & 31, wave = tid >> 5;
  int bid = blockIdx.x;
  int h = bid % NH;
  int c = (bid / NH) % NC_;
  int b = bid / (NH * NC_);
  long long base = (long long)b * LSEQ + c * CHUNK_;
  const float* acs = Acs + (((long long)b * NH + h) * NC_ + c) * CHUNK_;

  __shared__ float w[CHUNK_];
  if (tid < CHUNK_) {
    float last = acs[CHUNK_ - 1];
    w[tid] = dtsp[(base + tid) * NH + h] * expf(last - acs[tid]);
  }
  __syncthreads();

  int p0 = (wave >> 2) * 16, n0 = (wave & 3) * 16;
  int ml = lane & 15, hi = lane >> 4;
  v8f acc = (v8f)0.0f;
  for (int k0 = 0; k0 < CHUNK_; k0 += 4) {
    int ka = k0 + hi * 2;
    v2f a, bb;
    a.x = xs[(base + ka)     * DINNER + h * HDIM + p0 + ml] * w[ka];
    a.y = xs[(base + ka + 1) * DINNER + h * HDIM + p0 + ml] * w[ka + 1];
    bb.x = Bm[(base + ka)     * DSTATE + n0 + ml];
    bb.y = Bm[(base + ka + 1) * DSTATE + n0 + ml];
    acc = __builtin_amdgcn_wmma_f32_16x16x4_f32(
        false, a, false, bb, (short)0, acc, false, false);
  }
  float* st = states + (((long long)b * NC_ + c) * NH + h) * (HDIM * DSTATE);
  #pragma unroll
  for (int r = 0; r < 8; ++r)
    st[(p0 + hi * 8 + r) * DSTATE + n0 + ml] = acc[r];
}

// ---------------------------------------------------------------------------
// Sequential inter-chunk scan: prev[c] = carry; carry = carry*decay[c]+states[c]
// one block per (b,h), 256 threads * 16 elements = 4096-elem state
// ---------------------------------------------------------------------------
__global__ void ssd_scan_kernel(const float* __restrict__ states,
                                const float* __restrict__ cdec,
                                float* __restrict__ prev) {
  int tid = threadIdx.x;
  int bid = blockIdx.x;
  int h = bid % NH;
  int b = bid / NH;
  float carry[16];
  #pragma unroll
  for (int j = 0; j < 16; ++j) carry[j] = 0.0f;
  const float* cd = cdec + ((long long)b * NH + h) * NC_;
  for (int c = 0; c < NC_; ++c) {
    long long off = (((long long)b * NC_ + c) * NH + h) * (HDIM * DSTATE);
    float d = cd[c];
    #pragma unroll
    for (int j = 0; j < 16; ++j) {
      int e = tid + j * 256;
      float s = states[off + e];
      prev[off + e] = carry[j];
      carry[j] = carry[j] * d + s;
    }
  }
}

// ---------------------------------------------------------------------------
// Per-chunk output: Y = (tril(C B^T * segexp)) @ Xd + (C*exp(Acs)) @ prev^T + D*X
// CB tile built via f32 WMMA into LDS, masked, then two GEMM passes.
// ---------------------------------------------------------------------------
__global__ void __launch_bounds__(512) ssd_y_kernel(
    const float* __restrict__ xs, const float* __restrict__ Bm,
    const float* __restrict__ Cm, const float* __restrict__ dtsp,
    const float* __restrict__ Acs, const float* __restrict__ prev,
    const float* __restrict__ Dp, float* __restrict__ y) {
  __shared__ float CB[CHUNK_ * CHUNK_];
  __shared__ float acs_s[CHUNK_], wdt[CHUNK_], exl[CHUNK_];
  int tid = threadIdx.x, lane = tid & 31, wave = tid >> 5;
  int bid = blockIdx.x;
  int h = bid % NH;
  int c = (bid / NH) % NC_;
  int b = bid / (NH * NC_);
  long long base = (long long)b * LSEQ + c * CHUNK_;
  const float* acs = Acs + (((long long)b * NH + h) * NC_ + c) * CHUNK_;

  if (tid < CHUNK_) {
    float a = acs[tid];
    acs_s[tid] = a;
    exl[tid] = expf(a);
    wdt[tid] = dtsp[(base + tid) * NH + h];
  }
  __syncthreads();

  int ml = lane & 15, hi = lane >> 4;
  int t0 = (wave >> 2) * 16;   // l tile
  int t1 = (wave & 3) * 16;    // s / p tile

  // Phase A: CB[l,s] = sum_n C[l,n] B[s,n]
  v8f acc = (v8f)0.0f;
  for (int k0 = 0; k0 < DSTATE; k0 += 4) {
    int ka = k0 + hi * 2;
    v2f a, bb;
    a.x = Cm[(base + t0 + ml) * DSTATE + ka];
    a.y = Cm[(base + t0 + ml) * DSTATE + ka + 1];
    bb.x = Bm[(base + t1 + ml) * DSTATE + ka];
    bb.y = Bm[(base + t1 + ml) * DSTATE + ka + 1];
    acc = __builtin_amdgcn_wmma_f32_16x16x4_f32(
        false, a, false, bb, (short)0, acc, false, false);
  }
  #pragma unroll
  for (int r = 0; r < 8; ++r)
    CB[(t0 + hi * 8 + r) * CHUNK_ + t1 + ml] = acc[r];
  __syncthreads();

  // Phase B: causal mask + segsum exponent
  for (int e = tid; e < CHUNK_ * CHUNK_; e += 512) {
    int l = e >> 6, s = e & 63;
    float v = (s <= l) ? CB[e] * expf(acs_s[l] - acs_s[s]) : 0.0f;
    CB[e] = v;
  }
  __syncthreads();

  // Phase C: Y = M @ Xd  (+ off-diag)  [A from LDS -> ds_load traffic]
  acc = (v8f)0.0f;
  for (int k0 = 0; k0 < CHUNK_; k0 += 4) {
    int ka = k0 + hi * 2;
    v2f a, bb;
    a.x = CB[(t0 + ml) * CHUNK_ + ka];
    a.y = CB[(t0 + ml) * CHUNK_ + ka + 1];
    bb.x = xs[(base + ka)     * DINNER + h * HDIM + t1 + ml] * wdt[ka];
    bb.y = xs[(base + ka + 1) * DINNER + h * HDIM + t1 + ml] * wdt[ka + 1];
    acc = __builtin_amdgcn_wmma_f32_16x16x4_f32(
        false, a, false, bb, (short)0, acc, false, false);
  }
  const float* pv = prev + (((long long)b * NC_ + c) * NH + h) * (HDIM * DSTATE);
  for (int k0 = 0; k0 < DSTATE; k0 += 4) {
    int ka = k0 + hi * 2;
    v2f a, bb;
    float el = exl[t0 + ml];
    a.x = Cm[(base + t0 + ml) * DSTATE + ka] * el;
    a.y = Cm[(base + t0 + ml) * DSTATE + ka + 1] * el;
    bb.x = pv[(t1 + ml) * DSTATE + ka];
    bb.y = pv[(t1 + ml) * DSTATE + ka + 1];
    acc = __builtin_amdgcn_wmma_f32_16x16x4_f32(
        false, a, false, bb, (short)0, acc, false, false);
  }

  float dph = Dp[h];
  #pragma unroll
  for (int r = 0; r < 8; ++r) {
    long long row = base + t0 + hi * 8 + r;
    int p = t1 + ml;
    float xv = xs[row * DINNER + h * HDIM + p];
    y[row * DINNER + h * HDIM + p] = acc[r] + dph * xv;
  }
}

// ---------------------------------------------------------------------------
// u = y * silu(z); RMS-norm over 512; * norm_w; -> bf16 for out_proj
// ---------------------------------------------------------------------------
__global__ void rms_silu_kernel(const float* __restrict__ y, const float* __restrict__ zx,
                                const float* __restrict__ norm_w, bf16_t* __restrict__ normed) {
  int row = blockIdx.x, tid = threadIdx.x;  // 256 threads, 2 elems each
  __shared__ float red[256];
  long long yb = (long long)row * DINNER;
  long long zb = (long long)row * NPROJP;
  float z0 = zx[zb + tid];
  float z1 = zx[zb + tid + 256];
  float u0 = y[yb + tid]       * (z0 / (1.0f + expf(-z0)));
  float u1 = y[yb + tid + 256] * (z1 / (1.0f + expf(-z1)));
  red[tid] = u0 * u0 + u1 * u1;
  __syncthreads();
  for (int s = 128; s > 0; s >>= 1) {
    if (tid < s) red[tid] += red[tid + s];
    __syncthreads();
  }
  float r = rsqrtf(red[0] / (float)DINNER + 1e-12f);
  normed[yb + tid]       = (bf16_t)(u0 * r * norm_w[tid]);
  normed[yb + tid + 256] = (bf16_t)(u1 * r * norm_w[tid + 256]);
}

// ---------------------------------------------------------------------------
// LayerNorm(a + res) * g + b ; optional bf16 copy of output
// ---------------------------------------------------------------------------
template <bool WBF>
__global__ void ln_kernel(const float* __restrict__ a, const float* __restrict__ res,
                          const float* __restrict__ g, const float* __restrict__ bvec,
                          float* __restrict__ out, bf16_t* __restrict__ outbf) {
  int row = blockIdx.x, tid = threadIdx.x;  // 256 threads = DMODEL
  __shared__ float s1[256], s2[256];
  long long off = (long long)row * DMODEL + tid;
  float v = a[off] + res[off];
  s1[tid] = v;
  s2[tid] = v * v;
  __syncthreads();
  for (int s = 128; s > 0; s >>= 1) {
    if (tid < s) { s1[tid] += s1[tid + s]; s2[tid] += s2[tid + s]; }
    __syncthreads();
  }
  float mean = s1[0] * (1.0f / 256.0f);
  float var  = s2[0] * (1.0f / 256.0f) - mean * mean;
  float o = (v - mean) * rsqrtf(var + 1e-12f) * g[tid] + bvec[tid];
  out[off] = o;
  if (WBF) outbf[off] = (bf16_t)o;
}

// ---------------------------------------------------------------------------
// Workspace layout (256B aligned), with region reuse; peak ~916 MB
// ---------------------------------------------------------------------------
static constexpr size_t AL256(size_t x) { return (x + 255) & ~(size_t)255; }
static constexpr size_t OFF_XBF  = 0;
static constexpr size_t OFF_WIN  = OFF_XBF  + AL256((size_t)MROWS * DMODEL * 2);
static constexpr size_t OFF_WOUT = OFF_WIN  + AL256((size_t)NPROJP * DMODEL * 2);
static constexpr size_t OFF_WFC1 = OFF_WOUT + AL256((size_t)DMODEL * DINNER * 2);
static constexpr size_t OFF_WFC2 = OFF_WFC1 + AL256((size_t)DFFN * DMODEL * 2);
static constexpr size_t OFF_ZX   = OFF_WFC2 + AL256((size_t)DMODEL * DFFN * 2);
static constexpr size_t OFF_XS   = OFF_ZX   + AL256((size_t)MROWS * NPROJP * 4);
static constexpr size_t OFF_BM   = OFF_XS   + AL256((size_t)MROWS * DINNER * 4);
static constexpr size_t OFF_CM   = OFF_BM   + AL256((size_t)MROWS * DSTATE * 4);
static constexpr size_t OFF_DTSP = OFF_CM   + AL256((size_t)MROWS * DSTATE * 4);
static constexpr size_t OFF_ACS  = OFF_DTSP + AL256((size_t)MROWS * NH * 4);
static constexpr size_t OFF_CDEC = OFF_ACS  + AL256((size_t)BATCH * NH * NC_ * CHUNK_ * 4);
static constexpr size_t OFF_ST   = OFF_CDEC + AL256((size_t)BATCH * NH * NC_ * 4);
static constexpr size_t OFF_PREV = OFF_ST   + AL256((size_t)BATCH * NC_ * NH * HDIM * DSTATE * 4);
static constexpr size_t OFF_Y    = OFF_PREV + AL256((size_t)BATCH * NC_ * NH * HDIM * DSTATE * 4);
// reuse after SSD phase completes:
static constexpr size_t OFF_NRM  = OFF_PREV;                                   // bf16 [M,512]
static constexpr size_t OFF_OUT1 = OFF_ST;                                     // f32  [M,256]
static constexpr size_t OFF_HID  = OFF_ST + AL256((size_t)MROWS * DMODEL * 4); // f32  [M,256]
static constexpr size_t OFF_HIDB = OFF_Y;                                      // bf16 [M,256]
static constexpr size_t OFF_H1   = OFF_ZX;                                     // bf16 [M,1024]
static constexpr size_t OFF_H2   = OFF_XS;                                     // f32  [M,256]

extern "C" void kernel_launch(void* const* d_in, const int* in_sizes, int n_in,
                              void* d_out, int out_size, void* d_ws, size_t ws_size,
                              hipStream_t stream) {
  const float* x       = (const float*)d_in[0];
  const float* in_w    = (const float*)d_in[1];
  const float* in_b    = (const float*)d_in[2];
  const float* conv_w  = (const float*)d_in[3];
  const float* conv_b  = (const float*)d_in[4];
  const float* dt_bias = (const float*)d_in[5];
  const float* A_log   = (const float*)d_in[6];
  const float* Dp      = (const float*)d_in[7];
  const float* norm_w  = (const float*)d_in[8];
  const float* out_w   = (const float*)d_in[9];
  const float* out_b   = (const float*)d_in[10];
  const float* ln_g    = (const float*)d_in[11];
  const float* ln_b    = (const float*)d_in[12];
  const float* fc1_w   = (const float*)d_in[13];
  const float* fc1_b   = (const float*)d_in[14];
  const float* fc2_w   = (const float*)d_in[15];
  const float* fc2_b   = (const float*)d_in[16];
  const float* fln_g   = (const float*)d_in[17];
  const float* fln_b   = (const float*)d_in[18];

  char* ws = (char*)d_ws;
  bf16_t* x_bf   = (bf16_t*)(ws + OFF_XBF);
  bf16_t* w_in   = (bf16_t*)(ws + OFF_WIN);
  bf16_t* w_out  = (bf16_t*)(ws + OFF_WOUT);
  bf16_t* w_fc1  = (bf16_t*)(ws + OFF_WFC1);
  bf16_t* w_fc2  = (bf16_t*)(ws + OFF_WFC2);
  float*  zx     = (float*)(ws + OFF_ZX);
  float*  xs     = (float*)(ws + OFF_XS);
  float*  Bmat   = (float*)(ws + OFF_BM);
  float*  Cmat   = (float*)(ws + OFF_CM);
  float*  dtsp   = (float*)(ws + OFF_DTSP);
  float*  Acs    = (float*)(ws + OFF_ACS);
  float*  cdec   = (float*)(ws + OFF_CDEC);
  float*  states = (float*)(ws + OFF_ST);
  float*  prevs  = (float*)(ws + OFF_PREV);
  float*  ybuf   = (float*)(ws + OFF_Y);
  bf16_t* normed = (bf16_t*)(ws + OFF_NRM);
  float*  out1   = (float*)(ws + OFF_OUT1);
  float*  hidden = (float*)(ws + OFF_HID);
  bf16_t* hid_bf = (bf16_t*)(ws + OFF_HIDB);
  bf16_t* h1     = (bf16_t*)(ws + OFF_H1);
  float*  h2     = (float*)(ws + OFF_H2);
  float*  outp   = (float*)d_out;

  // --- pack fp32 -> bf16 (activations + padded weights) ---
  {
    long long t;
    t = (long long)MROWS * DMODEL;
    cvt_pad_kernel<<<(unsigned)((t + 255) / 256), 256, 0, stream>>>(x, x_bf, MROWS, DMODEL, t);
    t = (long long)NPROJP * DMODEL;
    cvt_pad_kernel<<<(unsigned)((t + 255) / 256), 256, 0, stream>>>(in_w, w_in, NPROJ, DMODEL, t);
    t = (long long)DMODEL * DINNER;
    cvt_pad_kernel<<<(unsigned)((t + 255) / 256), 256, 0, stream>>>(out_w, w_out, DMODEL, DINNER, t);
    t = (long long)DFFN * DMODEL;
    cvt_pad_kernel<<<(unsigned)((t + 255) / 256), 256, 0, stream>>>(fc1_w, w_fc1, DFFN, DMODEL, t);
    t = (long long)DMODEL * DFFN;
    cvt_pad_kernel<<<(unsigned)((t + 255) / 256), 256, 0, stream>>>(fc2_w, w_fc2, DMODEL, DFFN, t);
  }

  // --- in_proj GEMM: zxbcdt [M, 1160] (stride 1168) ---
  gemm_bf16_kernel<0, 0><<<dim3((NPROJP + 63) / 64, MROWS / 64), 256, 0, stream>>>(
      x_bf, w_in, in_b, zx, MROWS, NPROJP, NPROJ, DMODEL, NPROJP);

  // --- causal conv + silu + split ---
  {
    long long t = (long long)MROWS * CONVD;
    conv_silu_split_kernel<<<(unsigned)((t + 255) / 256), 256, 0, stream>>>(
        zx, conv_w, conv_b, xs, Bmat, Cmat);
  }

  // --- dt softplus + dA cumsum + chunk decay ---
  dt_scan_kernel<<<BATCH * NH * NC_, CHUNK_, 0, stream>>>(zx, dt_bias, A_log, dtsp, Acs, cdec);

  // --- per-chunk states (f32 WMMA) ---
  ssd_states_kernel<<<BATCH * NC_ * NH, 512, 0, stream>>>(xs, Bmat, dtsp, Acs, states);

  // --- inter-chunk sequential scan ---
  ssd_scan_kernel<<<BATCH * NH, 256, 0, stream>>>(states, cdec, prevs);

  // --- per-chunk Y (f32 WMMA + LDS) ---
  ssd_y_kernel<<<BATCH * NC_ * NH, 512, 0, stream>>>(
      xs, Bmat, Cmat, dtsp, Acs, prevs, Dp, ybuf);

  // --- gated RMS norm -> bf16 ---
  rms_silu_kernel<<<MROWS, 256, 0, stream>>>(ybuf, zx, norm_w, normed);

  // --- out_proj GEMM ---
  gemm_bf16_kernel<0, 0><<<dim3(DMODEL / 64, MROWS / 64), 256, 0, stream>>>(
      normed, w_out, out_b, out1, MROWS, DMODEL, DMODEL, DINNER, DMODEL);

  // --- LN1: hidden = LN(out1 + x) (f32 + bf16) ---
  ln_kernel<true><<<MROWS, DMODEL, 0, stream>>>(out1, x, ln_g, ln_b, hidden, hid_bf);

  // --- fc1 GEMM + hardswish -> bf16 ---
  gemm_bf16_kernel<1, 1><<<dim3(DFFN / 64, MROWS / 64), 256, 0, stream>>>(
      hid_bf, w_fc1, fc1_b, h1, MROWS, DFFN, DFFN, DMODEL, DFFN);

  // --- fc2 GEMM ---
  gemm_bf16_kernel<0, 0><<<dim3(DMODEL / 64, MROWS / 64), 256, 0, stream>>>(
      h1, w_fc2, fc2_b, h2, MROWS, DMODEL, DMODEL, DFFN, DMODEL);

  // --- LN2: out = LN(h2 + hidden) ---
  ln_kernel<false><<<MROWS, DMODEL, 0, stream>>>(h2, hidden, fln_g, fln_b, outp, nullptr);
}
